// SparseLinear_45741401702968
// MI455X (gfx1250) — matrix operand order repaired and checked by reading, not compiled, not run on hardware
//
#include <hip/hip_runtime.h>
#include <hip/hip_bf16.h>
#include <stdint.h>

#define OUTF   4096
#define INFEA  4096
#define NBATCH 2048

#define KTILE  64            // K depth staged per TDM transfer
#define APITCH 144           // LDS row pitch: 64*2 bytes + 16B TDM pad (bank spread)

typedef __attribute__((ext_vector_type(16))) __bf16 v16bf;
typedef __attribute__((ext_vector_type(8)))  __bf16 v8bf;
typedef __attribute__((ext_vector_type(8)))  float  v8f;

typedef __attribute__((ext_vector_type(4))) unsigned int tdm_v4u;
typedef __attribute__((ext_vector_type(8))) int          tdm_v8i;
typedef __attribute__((ext_vector_type(4))) int          tdm_v4i;

__device__ __forceinline__ unsigned short f32_to_bf16_rne(float f) {
    union { float f; unsigned int u; } x; x.f = f;
    unsigned int u = x.u;
    if ((u & 0x7fffffffu) > 0x7f800000u)            // NaN: quiet it
        return (unsigned short)((u >> 16) | 0x40);
    u += 0x7fffu + ((u >> 16) & 1u);                // round to nearest even
    return (unsigned short)(u >> 16);
}

// ---------------------------------------------------------------------------
// TDM 2-D tile load: global (row-major, elem=2B) -> LDS, with LDS row padding.
// Descriptor per CDNA5 ISA (D# groups 0/1; groups 2/3
// zero for a 2-D tensor). pad_interval=4 -> pad every 32 DWORDs (one 128B
// row); pad_amount=3 -> 4 DWORDs (16B) => LDS pitch 144B.
// ---------------------------------------------------------------------------
__device__ __forceinline__ void tdm_load_2d(unsigned lds_off,
                                            unsigned long long gaddr,
                                            unsigned tensor_d0,
                                            unsigned tensor_d1,
                                            unsigned stride0,
                                            unsigned tile_d0,
                                            unsigned tile_d1) {
    tdm_v4u g0;
    g0[0] = 1u;                                            // count=1, user D#
    g0[1] = lds_off;                                       // LDS byte address
    g0[2] = (unsigned)(gaddr & 0xFFFFFFFFull);             // global_addr[31:0]
    g0[3] = (unsigned)((gaddr >> 32) & 0x1FFFFFFull)       // global_addr[56:32]
          | (2u << 30);                                    // type=2 ("image")

    tdm_v8i g1;
    g1[0] = (int)((1u << 16)                               // data_size: 2 bytes
                | (1u << 20)                               // pad_enable
                | (4u << 22)                               // pad_interval: 32 DW
                | (3u << 25));                             // pad_amount: 4 DW
    g1[1] = (int)((tensor_d0 & 0xFFFFu) << 16);            // dim0[15:0] @ [63:48]
    g1[2] = (int)(((tensor_d0 >> 16) & 0xFFFFu)            // dim0[31:16]
                | ((tensor_d1 & 0xFFFFu) << 16));          // dim1[15:0]
    g1[3] = (int)(((tensor_d1 >> 16) & 0xFFFFu)            // dim1[31:16]
                | ((tile_d0 & 0xFFFFu) << 16));            // tile_dim0
    g1[4] = (int)(tile_d1 & 0xFFFFu);                      // tile_dim1 (dim2=0)
    g1[5] = (int)stride0;                                  // dim0_stride[31:0]
    g1[6] = 0;                                             // stride hi / dim1_stride lo
    g1[7] = 0;

    tdm_v4i z4 = {0, 0, 0, 0};
#if defined(__clang_major__) && __clang_major__ >= 23
    tdm_v8i z8 = {0, 0, 0, 0, 0, 0, 0, 0};
    __builtin_amdgcn_tensor_load_to_lds(g0, g1, z4, z4, z8, 0);
#else
    __builtin_amdgcn_tensor_load_to_lds(g0, g1, z4, z4, 0);
#endif
}

// ---------------------------------------------------------------------------
// 1) COO scatter-add into dense fp32 W  (W[r, c] += v, duplicates sum)
// ---------------------------------------------------------------------------
__global__ void coo_scatter_kernel(const int* __restrict__ rows,
                                   const int* __restrict__ cols,
                                   const float* __restrict__ vals,
                                   float* __restrict__ Wf, int nnz) {
    int i = blockIdx.x * blockDim.x + threadIdx.x;
    if (i < nnz) {
        atomicAdd(&Wf[(size_t)rows[i] * INFEA + cols[i]], vals[i]);
    }
}

// ---------------------------------------------------------------------------
// 2) Pack fp32 W -> bf16 W (row-major), 4 elements per thread
// ---------------------------------------------------------------------------
__global__ void pack_w_bf16_kernel(const float* __restrict__ src,
                                   unsigned short* __restrict__ dst, int n4) {
    int i = blockIdx.x * blockDim.x + threadIdx.x;
    if (i < n4) {
        float4 v = ((const float4*)src)[i];
        uint2 p;
        p.x = (unsigned int)f32_to_bf16_rne(v.x) |
              ((unsigned int)f32_to_bf16_rne(v.y) << 16);
        p.y = (unsigned int)f32_to_bf16_rne(v.z) |
              ((unsigned int)f32_to_bf16_rne(v.w) << 16);
        ((uint2*)dst)[i] = p;
    }
}

// ---------------------------------------------------------------------------
// 3) Transpose+pack inp [IN, BATCH] fp32 -> inpT [BATCH, IN] bf16 (LDS tiles)
// ---------------------------------------------------------------------------
__global__ __launch_bounds__(256)
void transpose_pack_kernel(const float* __restrict__ inp,
                           unsigned short* __restrict__ inpT) {
    __shared__ float tile[32][33];
    const int b0 = blockIdx.x * 32;
    const int k0 = blockIdx.y * 32;
    const int tx = threadIdx.x;       // 0..31
    const int ty = threadIdx.y;       // 0..7
    #pragma unroll
    for (int i = 0; i < 32; i += 8)
        tile[ty + i][tx] = inp[(size_t)(k0 + ty + i) * NBATCH + b0 + tx];
    __syncthreads();
    #pragma unroll
    for (int i = 0; i < 32; i += 8)
        inpT[(size_t)(b0 + ty + i) * INFEA + k0 + tx] =
            f32_to_bf16_rne(tile[tx][ty + i]);
}

// ---------------------------------------------------------------------------
// 4) TDM-fed WMMA GEMM: out[m,n] = sum_k Wb[m,k]*Xb[n,k] + bias[m]
//    Block = 8 waves (2 along M x 4 along N) -> 128x256 block tile.
//    Wave tile = 64x64 = 4x4 WMMA tiles -> 16 WMMAs per 16 ds_load_b128 per
//    K-step of 32 (A/B fragments each reused 4x). Wave 0 drives
//    double-buffered TDM K-panels (A: 128x64, B: 256x64 bf16, pitch 144B).
// ---------------------------------------------------------------------------
__global__ __launch_bounds__(256)
void wmma_gemm_tdm_kernel(const unsigned short* __restrict__ Wb,
                          const unsigned short* __restrict__ Xb,
                          const float* __restrict__ bias,
                          float* __restrict__ out) {
    __shared__ __align__(16) unsigned char shA[2][128 * APITCH];   // 2 x 18 KB
    __shared__ __align__(16) unsigned char shB[2][256 * APITCH];   // 2 x 36 KB

    const int lane = threadIdx.x & 31;
    const int wid  = threadIdx.x >> 5;
    const int half = lane >> 4;   // K-half select per 16-bit WMMA layouts
    const int l    = lane & 15;   // row (A) / column (B,C,D) index in tile

    const int wm = wid & 1;       // 2 waves along M
    const int wn = wid >> 1;      // 4 waves along N
    const int m_block = blockIdx.x * 128;
    const int n_block = blockIdx.y * 256;

    v8f acc[4][4];
    #pragma unroll
    for (int mi = 0; mi < 4; ++mi)
        #pragma unroll
        for (int ni = 0; ni < 4; ++ni)
            acc[mi][ni] = (v8f){0.f, 0.f, 0.f, 0.f, 0.f, 0.f, 0.f, 0.f};

    const unsigned long long Wg =
        (unsigned long long)(uintptr_t)Wb + (unsigned long long)m_block * (INFEA * 2);
    const unsigned long long Xg =
        (unsigned long long)(uintptr_t)Xb + (unsigned long long)n_block * (INFEA * 2);

    const int T = INFEA / KTILE;   // 64 K-tiles

    if (wid == 0) {
        tdm_load_2d((unsigned)(uintptr_t)&shA[0][0], Wg, INFEA, OUTF,   INFEA, KTILE, 128);
        tdm_load_2d((unsigned)(uintptr_t)&shB[0][0], Xg, INFEA, NBATCH, INFEA, KTILE, 256);
    }

    for (int kt = 0; kt < T; ++kt) {
        const int cur = kt & 1;
        if (wid == 0) {
            if (kt + 1 < T) {
                const unsigned long long koff =
                    (unsigned long long)(kt + 1) * (KTILE * 2);
                tdm_load_2d((unsigned)(uintptr_t)&shA[cur ^ 1][0], Wg + koff,
                            INFEA, OUTF, INFEA, KTILE, 128);
                tdm_load_2d((unsigned)(uintptr_t)&shB[cur ^ 1][0], Xg + koff,
                            INFEA, NBATCH, INFEA, KTILE, 256);
                __builtin_amdgcn_s_wait_tensorcnt(2);  // current tile landed
            } else {
                __builtin_amdgcn_s_wait_tensorcnt(0);
            }
        }
        __syncthreads();   // release all waves to read buffer `cur`

        const unsigned char* A = &shA[cur][0];
        const unsigned char* B = &shB[cur][0];

        #pragma unroll
        for (int kk = 0; kk < KTILE; kk += 32) {
            // A fragment: lane l = row l; half 0 -> K {0..7,16..23},
            // half 1 -> K {8..15,24..31}  (ISA 7.12.2, 16-bit A 16x32)
            v16bf a[4];
            #pragma unroll
            for (int mi = 0; mi < 4; ++mi) {
                const unsigned char* ar = A + (wm * 64 + mi * 16 + l) * APITCH;
                v8bf lo = *(const v8bf*)(const void*)(ar + (kk + half * 8) * 2);
                v8bf hi = *(const v8bf*)(const void*)(ar + (kk + 16 + half * 8) * 2);
                #pragma unroll
                for (int e = 0; e < 8; ++e) { a[mi][e] = lo[e]; a[mi][e + 8] = hi[e]; }
            }
            // B fragment: lane l = column l; half 0 -> K {0..7,16..23}, half 1 -> K {8..15,24..31}
            v16bf b[4];
            #pragma unroll
            for (int ni = 0; ni < 4; ++ni) {
                const unsigned char* br = B + (wn * 64 + ni * 16 + l) * APITCH;
                v8bf lo = *(const v8bf*)(const void*)(br + (kk + half * 8) * 2);
                v8bf hi = *(const v8bf*)(const void*)(br + (kk + 16 + half * 8) * 2);
                #pragma unroll
                for (int e = 0; e < 8; ++e) { b[ni][e] = lo[e]; b[ni][e + 8] = hi[e]; }
            }
            #pragma unroll
            for (int mi = 0; mi < 4; ++mi)
                #pragma unroll
                for (int ni = 0; ni < 4; ++ni)
                    acc[mi][ni] = __builtin_amdgcn_wmma_f32_16x16x32_bf16(
                        false, a[mi], false, b[ni],
                        (short)0, acc[mi][ni], false, false);
        }
        __syncthreads();   // all reads of buffer `cur` done before overwrite
    }

    // C/D layout: VGPR i -> M = i + 8*half, N = l  (ISA 7.12.2, 32-bit C/D)
    #pragma unroll
    for (int mi = 0; mi < 4; ++mi) {
        #pragma unroll
        for (int i = 0; i < 8; ++i) {
            const int row = m_block + wm * 64 + mi * 16 + half * 8 + i;
            const float bv = bias[row];
            #pragma unroll
            for (int ni = 0; ni < 4; ++ni)
                out[(size_t)row * NBATCH + n_block + wn * 64 + ni * 16 + l] =
                    acc[mi][ni][i] + bv;
        }
    }
}

// ---------------------------------------------------------------------------
extern "C" void kernel_launch(void* const* d_in, const int* in_sizes, int n_in,
                              void* d_out, int out_size, void* d_ws, size_t ws_size,
                              hipStream_t stream) {
    const float* inp  = (const float*)d_in[0];
    const float* vals = (const float*)d_in[1];
    const float* bias = (const float*)d_in[2];
    const int*   rows = (const int*)d_in[3];
    const int*   cols = (const int*)d_in[4];
    const int    nnz  = in_sizes[3];

    char* ws = (char*)d_ws;
    float*          Wf = (float*)ws;                                       // 64 MB
    unsigned short* Wb = (unsigned short*)(ws + (size_t)OUTF * INFEA * 4); // +32 MB
    unsigned short* Xb = (unsigned short*)(ws + (size_t)OUTF * INFEA * 6); // +16 MB
    float* out = (float*)d_out;

    // 1) zero dense W, scatter-add COO values
    hipMemsetAsync(Wf, 0, (size_t)OUTF * INFEA * sizeof(float), stream);
    coo_scatter_kernel<<<(nnz + 255) / 256, 256, 0, stream>>>(rows, cols, vals, Wf, nnz);

    // 2) pack W -> bf16
    const int n4 = OUTF * INFEA / 4;
    pack_w_bf16_kernel<<<(n4 + 255) / 256, 256, 0, stream>>>(Wf, Wb, n4);

    // 3) transpose+pack inp -> [BATCH][IN] bf16
    dim3 tg(NBATCH / 32, INFEA / 32), tb(32, 8);
    transpose_pack_kernel<<<tg, tb, 0, stream>>>(inp, Xb);

    // 4) TDM-fed WMMA GEMM + bias
    dim3 gg(OUTF / 128, NBATCH / 256);
    wmma_gemm_tdm_kernel<<<gg, 256, 0, stream>>>(Wb, Xb, bias, out);
}
